// MDJCM_Model_75909251989602
// MI455X (gfx1250) — matrix operand order repaired and checked by reading, not compiled, not run on hardware
//
#include <hip/hip_runtime.h>
#include <hip/hip_bf16.h>
#include <math.h>

// ---------------------------------------------------------------------------
// MDJCM model decode path for MI455X (gfx1250, wave32, WMMA + async-to-LDS).
// Shapes fixed by setup_inputs(): B=16, C=256, h=32, w=48, block_len=128,
// dff=512, modulation_order=16, snr=10.
// ---------------------------------------------------------------------------

typedef __attribute__((ext_vector_type(8)))  float          v8f;
typedef __attribute__((ext_vector_type(16))) __bf16         v16bf;
typedef __attribute__((ext_vector_type(16))) unsigned short v16us;

#define BB          16
#define NPER        393216            // C*h*w = 256*32*48
#define TOTALN      (BB * NPER)       // 6291456
#define BLKLEN      128
#define DFF         512
#define NBLOCKS     (TOTALN / BLKLEN) // 49152
#define CHUNK       1024
#define CH_PER_ROW  (NPER / CHUNK)    // 384
#define NCHUNK      (BB * CH_PER_ROW) // 6144
#define MLP_WAVES   4
#define NTILES      (NBLOCKS / 16)    // 3072
#define WITER       8192              // ushorts of packed weights per s-iteration (16 KB)

// workspace byte offsets
#define OFF_SUM   0                   // float   running sum of x^2*m
#define OFF_TOT   8                   // uint    running count of mask
#define OFF_CCNT  1024                // int[NCHUNK]   per-chunk mask counts
#define OFF_COFF  32768               // int[NCHUNK]   per-chunk exclusive offsets
#define OFF_WC    65536               // ushort[16*WITER] combined bf16 frag-packed weights
#define OFF_PACK  327680              // float[TOTALN] packed symbols / in-place dec

// modulation constants (modulation_order=16, snr=10 dB, power=1)
#define D_STEP   0.31622776601683794f   // sqrt(3/(2*16-2))
#define LO_CLIP  (-3.0f * D_STEP)       // (1-sqrt(16))*d
#define HI_CLIP  ( 3.0f * D_STEP)       // (sqrt(16)-1)*d
#define SIGMA    0.22360679774997896f   // sqrt(1/(2*10^(10/10)))

// ---------------------------------------------------------------------------
__device__ __forceinline__ unsigned short f2bf(float f) {
    unsigned int u = __float_as_uint(f);
    u += 0x7FFFu + ((u >> 16) & 1u);    // round-to-nearest-even
    return (unsigned short)(u >> 16);
}

__device__ __forceinline__ unsigned int pcg_hash(unsigned int v) {
    v = v * 747796405u + 2891336453u;
    unsigned int w = ((v >> ((v >> 28u) + 4u)) ^ v) * 277803737u;
    return (w >> 22u) ^ w;
}
__device__ __forceinline__ float u01(unsigned int idx, unsigned int s) {
    return pcg_hash(idx ^ (s * 0x9E3779B9u)) * 2.3283064365386963e-10f;
}

union Frag16 {
    uint4          q[2];
    v16us          v;
    unsigned short s[16];
};

__device__ __forceinline__ v8f wmma_bf16(v16us a, v16us b, v8f c) {
    return __builtin_amdgcn_wmma_f32_16x16x32_bf16(
        false, __builtin_bit_cast(v16bf, a),
        false, __builtin_bit_cast(v16bf, b),
        (short)0, c, false, false);
}

// ---------------------------------------------------------------------------
// K0: zero packed buffer + scalar accumulators (ws is poisoned; rerun safe).
// ---------------------------------------------------------------------------
__global__ __launch_bounds__(256) void k_init(float* packed, float* sumPwr,
                                              unsigned int* total) {
    int i      = blockIdx.x * blockDim.x + threadIdx.x;
    int stride = gridDim.x * blockDim.x;
    for (int j = i; j < TOTALN; j += stride) packed[j] = 0.0f;
    if (i == 0) { *sumPwr = 0.0f; *total = 0u; }
}

// ---------------------------------------------------------------------------
// K1: per-chunk mask count + global masked power reduction.
// ---------------------------------------------------------------------------
__global__ __launch_bounds__(256) void k_stats(const float* __restrict__ x,
                                               const unsigned char* __restrict__ mask,
                                               float* sumPwr, unsigned int* total,
                                               int* __restrict__ chunkCnt) {
    __shared__ float sf[256];
    __shared__ int   si[256];
    int c    = blockIdx.x;
    int tid  = threadIdx.x;
    int base = c * CHUNK + tid * 4;

    float s = 0.0f;
    int   n = 0;
#pragma unroll
    for (int j = 0; j < 4; ++j) {
        float xv = x[base + j];
        int   m  = mask[base + j] ? 1 : 0;
        s += xv * xv * (float)m;
        n += m;
    }
    sf[tid] = s; si[tid] = n;
    __syncthreads();
    for (int off = 128; off > 0; off >>= 1) {
        if (tid < off) { sf[tid] += sf[tid + off]; si[tid] += si[tid + off]; }
        __syncthreads();
    }
    if (tid == 0) {
        atomicAdd(sumPwr, sf[0]);
        atomicAdd(total, (unsigned int)si[0]);
        chunkCnt[c] = si[0];
    }
}

// ---------------------------------------------------------------------------
// K2: per-row exclusive scan over CH_PER_ROW chunk counts (16 rows).
// ---------------------------------------------------------------------------
__global__ __launch_bounds__(256) void k_scan(const int* __restrict__ cnt,
                                              int* __restrict__ offs) {
    __shared__ int sd[256];
    __shared__ int carry;
    int row = blockIdx.x, tid = threadIdx.x;
    if (tid == 0) carry = 0;
    __syncthreads();
    for (int base = 0; base < CH_PER_ROW; base += 256) {
        int i = base + tid;
        int v = (i < CH_PER_ROW) ? cnt[row * CH_PER_ROW + i] : 0;
        sd[tid] = v;
        __syncthreads();
        for (int off = 1; off < 256; off <<= 1) {
            int t = (tid >= off) ? sd[tid - off] : 0;
            __syncthreads();
            sd[tid] += t;
            __syncthreads();
        }
        if (i < CH_PER_ROW) offs[row * CH_PER_ROW + i] = sd[tid] - v + carry;
        int tot = sd[255];
        __syncthreads();
        if (tid == 0) carry += tot;
        __syncthreads();
    }
}

// ---------------------------------------------------------------------------
// K3a/K3b: repack weights into bf16 WMMA B-fragment layout, grouped so each
// s-iteration's 16 KB (8 W1 frags then 8 W2 frags, 1 KB each) is contiguous.
// B-frag (32x16): lane = N (lane&15), elements e = K 8*(lane>>4)+(e&7)+16*(e>>3).
//   iteration s region (ushort units): s*WITER
//     W1 frag (half,kt): + (half*4+kt)*512 + lane*16
//     W2 frag (nt2):     + 4096 + nt2*512 + lane*16
// ---------------------------------------------------------------------------
__global__ __launch_bounds__(256) void k_prep_w1(const float* __restrict__ W1,
                                                 unsigned short* __restrict__ wc) {
    int t = blockIdx.x * blockDim.x + threadIdx.x;   // 128 frags * 32 lanes
    if (t >= 128 * 32) return;
    int frag = t >> 5, lane = t & 31;
    int kt = frag >> 5, nt = frag & 31;              // frag = kt*32 + nt
    int s = nt >> 1, half = nt & 1;
    const float* src = W1 + (size_t)(kt * 32 + (lane >> 4) * 8) * DFF + nt * 16 + (lane & 15);
    unsigned short* dst = wc + (size_t)s * WITER + (half * 4 + kt) * 512 + lane * 16;
#pragma unroll
    for (int e = 0; e < 16; ++e) dst[e] = f2bf(src[(size_t)((e & 7) + (e >> 3) * 16) * DFF]);
}

__global__ __launch_bounds__(256) void k_prep_w2(const float* __restrict__ W2,
                                                 unsigned short* __restrict__ wc) {
    int t = blockIdx.x * blockDim.x + threadIdx.x;   // 128 frags * 32 lanes
    if (t >= 128 * 32) return;
    int frag = t >> 5, lane = t & 31;
    int s = frag >> 3, nt2 = frag & 7;               // frag = s*8 + nt2
    const float* src = W2 + (size_t)(s * 32 + (lane >> 4) * 8) * BLKLEN + nt2 * 16 + (lane & 15);
    unsigned short* dst = wc + (size_t)s * WITER + 4096 + nt2 * 512 + lane * 16;
#pragma unroll
    for (int e = 0; e < 16; ++e) dst[e] = f2bf(src[(size_t)((e & 7) + (e >> 3) * 16) * BLKLEN]);
}

// ---------------------------------------------------------------------------
// K4: modulate + dither + AWGN + ragged pack (left-compaction per row).
// ---------------------------------------------------------------------------
__global__ __launch_bounds__(256) void k_pack(const float* __restrict__ x,
                                              const unsigned char* __restrict__ mask,
                                              const float* sumPwr,
                                              const unsigned int* total,
                                              const int* __restrict__ chunkOff,
                                              float* __restrict__ packed) {
    __shared__ int sd[256];
    int c   = blockIdx.x;
    int tid = threadIdx.x;
    int row = c / CH_PER_ROW;
    int base = c * CHUNK + tid * 4;

    float avg   = sumPwr[0] / (float)total[0];
    float scale = 1.0f / sqrtf(avg * 2.0f);

    int m[4];
#pragma unroll
    for (int j = 0; j < 4; ++j) m[j] = mask[base + j] ? 1 : 0;
    int tsum = m[0] + m[1] + m[2] + m[3];

    sd[tid] = tsum;
    __syncthreads();
    for (int off = 1; off < 256; off <<= 1) {
        int t = (tid >= off) ? sd[tid - off] : 0;
        __syncthreads();
        sd[tid] += t;
        __syncthreads();
    }
    int excl = sd[tid] - tsum + chunkOff[c];

    int run = 0;
#pragma unroll
    for (int j = 0; j < 4; ++j) {
        if (m[j]) {
            unsigned int idx = (unsigned int)(base + j);
            float xm = x[base + j] * scale;
            xm = fminf(fmaxf(xm, LO_CLIP), HI_CLIP);
            float u1 = (u01(idx, 1u) - 0.5f) * 2.0f * D_STEP;
            float u2 = (u01(idx, 2u) - 0.5f) * 2.0f * D_STEP;
            float r1 = fmaxf(u01(idx, 3u), 1e-10f);
            float r2 = u01(idx, 4u);
            float gn = sqrtf(-2.0f * logf(r1)) * cosf(6.28318530717958647f * r2);
            float yo = xm + u1 + gn * SIGMA + u2;
            packed[row * NPER + excl + run] = yo;
            ++run;
        }
    }
}

// ---------------------------------------------------------------------------
// K5: fused residual block-MLP on WMMA, in-place over `packed`.
// Per wave: one 16-row tile.  Weights are double-buffer staged into LDS with
// async global->LDS copies (whole WG shares them, 4x less fetch traffic, and
// the fetch of iteration s+1 overlaps the WMMAs of iteration s).
// ---------------------------------------------------------------------------
__global__ __launch_bounds__(128) void k_mlp(float* __restrict__ packed,
                                             const unsigned short* __restrict__ wc,
                                             const float* __restrict__ b1,
                                             const float* __restrict__ b2) {
    __shared__ __align__(64) unsigned short Wst[2][WITER];          // 2 x 16 KB
    __shared__ __align__(64) unsigned short Hst[MLP_WAVES][16][32]; // 4 KB

    int tid  = threadIdx.x;
    int wv   = tid >> 5;
    int lane = tid & 31;
    int tile = blockIdx.x * MLP_WAVES + wv;   // grid is exactly NTILES/MLP_WAVES

    float* X   = packed + (size_t)tile * 16 * BLKLEN;
    int   rlo  = lane & 15;
    int   hi16 = (lane & 16) ? 1 : 0;

    // ---- cooperative async stage of one 16 KB weight iteration ------------
    auto stage = [&](int buf, int s) {
        unsigned int ldsbase =
            (unsigned int)(uintptr_t)&Wst[buf][0];
        unsigned long long gbase =
            (unsigned long long)(uintptr_t)(wc + (size_t)s * WITER);
#pragma unroll
        for (int r = 0; r < 8; ++r) {
            unsigned int       lds = ldsbase + (unsigned int)((tid + r * 128) * 16);
            unsigned long long ga  = gbase + (unsigned long long)((tid + r * 128) * 16);
            asm volatile("global_load_async_to_lds_b128 %0, %1, off"
                         :: "v"(lds), "v"(ga) : "memory");
        }
    };

    // ---- X as four bf16 A-fragments (16x32 each) --------------------------
    v16us xa[4];
#pragma unroll
    for (int kt = 0; kt < 4; ++kt) {
        int b0 = kt * 32 + (hi16 ? 8 : 0);
        const float* px = X + rlo * BLKLEN + b0;
        Frag16 u;
#pragma unroll
        for (int e = 0; e < 8; ++e) u.s[e]     = f2bf(px[e]);
#pragma unroll
        for (int e = 0; e < 8; ++e) u.s[8 + e] = f2bf(px[16 + e]);
        xa[kt] = u.v;
    }

    // ---- D accumulators: residual X + b2 (C-layout) -----------------------
    v8f Dacc[8];
#pragma unroll
    for (int nt2 = 0; nt2 < 8; ++nt2) {
        int   col = nt2 * 16 + rlo;
        float bb  = b2[col];
#pragma unroll
        for (int g = 0; g < 8; ++g)
            Dacc[nt2][g] = X[(g + hi16 * 8) * BLKLEN + col] + bb;
    }

    stage(0, 0);   // prime the pipeline

    // ---- fused GEMM1 -> ReLU -> (LDS transpose) -> GEMM2 ------------------
    for (int s = 0; s < 16; ++s) {
        int cur = s & 1;
        // my wave's async copies (current buffer) complete, then WG-wide sync
        asm volatile("s_wait_asynccnt 0x0" ::: "memory");
        __syncthreads();
        if (s < 15) stage(cur ^ 1, s + 1);   // overlap next fetch with compute

        const unsigned short* W = &Wst[cur][0];

#pragma unroll
        for (int half = 0; half < 2; ++half) {
            int   nt = s * 2 + half;
            float bv = b1[nt * 16 + rlo];
            v8f acc;
#pragma unroll
            for (int g = 0; g < 8; ++g) acc[g] = bv;
#pragma unroll
            for (int kt = 0; kt < 4; ++kt) {
                Frag16 w;
                const uint4* wp =
                    (const uint4*)(W + (size_t)(half * 4 + kt) * 512 + lane * 16);
                w.q[0] = wp[0];
                w.q[1] = wp[1];
                acc = wmma_bf16(xa[kt], w.v, acc);
            }
#pragma unroll
            for (int g = 0; g < 8; ++g) {
                float v = acc[g];
                v = v > 0.0f ? v : 0.0f;                      // ReLU
                Hst[wv][g + hi16 * 8][half * 16 + rlo] = f2bf(v);
            }
        }
        // same-wave cross-lane LDS RAW: make the H staging visible
        asm volatile("s_wait_dscnt 0x0" ::: "memory");

        Frag16 a2;
        {
            int c0 = hi16 ? 8 : 0;
            const uint4* q0 = (const uint4*)&Hst[wv][rlo][c0];
            const uint4* q1 = (const uint4*)&Hst[wv][rlo][c0 + 16];
            a2.q[0] = q0[0];
            a2.q[1] = q1[0];
        }
#pragma unroll
        for (int nt2 = 0; nt2 < 8; ++nt2) {
            Frag16 w;
            const uint4* wp =
                (const uint4*)(W + 4096 + (size_t)nt2 * 512 + lane * 16);
            w.q[0] = wp[0];
            w.q[1] = wp[1];
            Dacc[nt2] = wmma_bf16(a2.v, w.v, Dacc[nt2]);
        }
        asm volatile("s_wait_dscnt 0x0" ::: "memory");  // WAR guard before restage
        __syncthreads();   // all waves done reading Wst[cur] before it is restaged
    }

    // ---- store dec in-place ----------------------------------------------
#pragma unroll
    for (int nt2 = 0; nt2 < 8; ++nt2) {
        int col = nt2 * 16 + rlo;
#pragma unroll
        for (int g = 0; g < 8; ++g)
            X[(g + hi16 * 8) * BLKLEN + col] = Dacc[nt2][g];
    }
}

// ---------------------------------------------------------------------------
// K6: scatter decoded symbols back to BCHW via recomputed pack positions.
// ---------------------------------------------------------------------------
__global__ __launch_bounds__(256) void k_gather(const unsigned char* __restrict__ mask,
                                                const int* __restrict__ chunkOff,
                                                const float* __restrict__ dec,
                                                float* __restrict__ out) {
    __shared__ int sd[256];
    int c    = blockIdx.x;
    int tid  = threadIdx.x;
    int row  = c / CH_PER_ROW;
    int base = c * CHUNK + tid * 4;

    int m[4];
#pragma unroll
    for (int j = 0; j < 4; ++j) m[j] = mask[base + j] ? 1 : 0;
    int tsum = m[0] + m[1] + m[2] + m[3];

    sd[tid] = tsum;
    __syncthreads();
    for (int off = 1; off < 256; off <<= 1) {
        int t = (tid >= off) ? sd[tid - off] : 0;
        __syncthreads();
        sd[tid] += t;
        __syncthreads();
    }
    int excl = sd[tid] - tsum + chunkOff[c];

    int run = 0;
#pragma unroll
    for (int j = 0; j < 4; ++j) {
        float v = 0.0f;
        if (m[j]) { v = dec[row * NPER + excl + run]; ++run; }
        out[base + j] = v;
    }
}

// ---------------------------------------------------------------------------
extern "C" void kernel_launch(void* const* d_in, const int* in_sizes, int n_in,
                              void* d_out, int out_size, void* d_ws, size_t ws_size,
                              hipStream_t stream) {
    (void)in_sizes; (void)n_in; (void)out_size; (void)ws_size;

    const float*         x    = (const float*)d_in[0];          // s_masked
    const float*         W1   = (const float*)d_in[1];          // (128,512)
    const float*         b1   = (const float*)d_in[2];          // (512,)
    const float*         W2   = (const float*)d_in[3];          // (512,128)
    const float*         b2   = (const float*)d_in[4];          // (128,)
    const unsigned char* mask = (const unsigned char*)d_in[5];  // bool (1 byte)
    // d_in[6..8] = modulation_order/snr/block_len: fixed by setup, folded in.

    char* ws = (char*)d_ws;
    float*          sumPwr   = (float*)(ws + OFF_SUM);
    unsigned int*   total    = (unsigned int*)(ws + OFF_TOT);
    int*            chunkCnt = (int*)(ws + OFF_CCNT);
    int*            chunkOff = (int*)(ws + OFF_COFF);
    unsigned short* wc       = (unsigned short*)(ws + OFF_WC);
    float*          packed   = (float*)(ws + OFF_PACK);
    float*          out      = (float*)d_out;

    k_init   <<<2048, 256, 0, stream>>>(packed, sumPwr, total);
    k_stats  <<<NCHUNK, 256, 0, stream>>>(x, mask, sumPwr, total, chunkCnt);
    k_scan   <<<BB, 256, 0, stream>>>(chunkCnt, chunkOff);
    k_prep_w1<<<16, 256, 0, stream>>>(W1, wc);
    k_prep_w2<<<16, 256, 0, stream>>>(W2, wc);
    k_pack   <<<NCHUNK, 256, 0, stream>>>(x, mask, sumPwr, total, chunkOff, packed);
    k_mlp    <<<NTILES / MLP_WAVES, MLP_WAVES * 32, 0, stream>>>(packed, wc, b1, b2);
    k_gather <<<NCHUNK, 256, 0, stream>>>(mask, chunkOff, packed, out);
}
